// LogAttentionTransformerLayer_52931176956033
// MI455X (gfx1250) — compile-verified, matrix operand order checked by reading, not run on hardware
//
#include <hip/hip_runtime.h>
#include <math.h>

// ---------------------------------------------------------------------------
// CDNA5 (gfx1250) WMMA types & helpers
// ---------------------------------------------------------------------------
typedef __bf16 v16bf __attribute__((ext_vector_type(16)));
typedef __bf16 bf8v  __attribute__((ext_vector_type(8)));
typedef __bf16 bf4v  __attribute__((ext_vector_type(4)));
typedef float  v8f   __attribute__((ext_vector_type(8)));
typedef float  f4    __attribute__((ext_vector_type(4)));

__device__ __forceinline__ v8f wmma_bf16(v16bf a, v16bf b, v8f c) {
  // D = A(16x32 bf16) * B(32x16 bf16) + C(16x16 f32)
  return __builtin_amdgcn_wmma_f32_16x16x32_bf16(
      /*neg_a=*/false, a, /*neg_b=*/false, b,
      /*c_mod=*/(short)0, c, /*reuse_a=*/false, /*reuse_b=*/false);
}

// 16x32 bf16 A-fragment (or 32x16 B-fragment stored [n][k]) from a row-major
// buffer (stride in elements).  ISA 7.12.2: lane l holds row (l&15);
// lanes 0-15: K 0..7 & 16..23;  lanes 16-31: K 8..15 & 24..31.
__device__ __forceinline__ v16bf load_frag(const __bf16* base, int stride) {
  const int l   = threadIdx.x & 31;
  const int row = l & 15;
  const int k0  = (l >> 4) << 3;   // 0 or 8
  const __bf16* p = base + row * stride + k0;
  bf8v lo = *(const bf8v*)(p);
  bf8v hi = *(const bf8v*)(p + 16);
  v16bf r;
#pragma unroll
  for (int i = 0; i < 8; ++i) { r[i] = lo[i]; r[i + 8] = hi[i]; }
  return r;
}

__device__ __forceinline__ v8f zero8() {
  v8f z = {0.f, 0.f, 0.f, 0.f, 0.f, 0.f, 0.f, 0.f};
  return z;
}

// ---------------------------------------------------------------------------
// Vectorized elementwise converts (16B global loads / stores)
// ---------------------------------------------------------------------------
__global__ void cvt_f32_bf16(const float* __restrict__ in,
                             __bf16* __restrict__ out, size_t n8) {
  for (size_t i = blockIdx.x * (size_t)blockDim.x + threadIdx.x; i < n8;
       i += (size_t)gridDim.x * blockDim.x) {
    const f4* p = (const f4*)(in + i * 8);
    f4 x0 = p[0], x1 = p[1];
    bf8v o;
#pragma unroll
    for (int j = 0; j < 4; ++j) { o[j] = (__bf16)x0[j]; o[j + 4] = (__bf16)x1[j]; }
    *(bf8v*)(out + i * 8) = o;
  }
}

// mean over P=4 of param_vectors [B,S,4,768] -> bf16 [B*S, 768]
__global__ void pv_mean_bf16(const float* __restrict__ pv,
                             __bf16* __restrict__ out) {
  const int C = 768;
  const size_t nch = (size_t)4096 * (C / 8);
  for (size_t i = blockIdx.x * (size_t)blockDim.x + threadIdx.x; i < nch;
       i += (size_t)gridDim.x * blockDim.x) {
    size_t t = i / 96;
    int c = (int)(i - t * 96) * 8;
    const float* p = pv + (t * 4) * C + c;
    f4 s0 = {0.f, 0.f, 0.f, 0.f}, s1 = s0;
#pragma unroll
    for (int pp = 0; pp < 4; ++pp) {
      const f4* q = (const f4*)(p + (size_t)pp * C);
      s0 += q[0];
      s1 += q[1];
    }
    bf8v o;
#pragma unroll
    for (int j = 0; j < 4; ++j) {
      o[j]     = (__bf16)(0.25f * s0[j]);
      o[j + 4] = (__bf16)(0.25f * s1[j]);
    }
    *(bf8v*)(out + t * C + c) = o;
  }
}

// ---------------------------------------------------------------------------
// C = A * W^T (+bias, optional exact-GELU) WMMA GEMM, 128x128 block tile.
// A: [M,K] bf16 row-major.  W: [N,K] bf16 row-major (B frag [n][k] == W rows).
// 256 threads = 8 waves arranged 4(M) x 2(N); each wave owns a 32x64 C tile
// (8 WMMAs per 32-K step against 6 fragment loads).  Double-buffered LDS +
// register prefetch + global_prefetch of the K+64 slice: one __syncthreads
// per K step, global traffic overlaps WMMA.
// EPI==1: v = gelu(v+bias) -> bf16.   EPI==2: v = v+bias -> f32.
// ---------------------------------------------------------------------------
template <int EPI>
__global__ __launch_bounds__(256) void gemm_bf16_wmma(
    const __bf16* __restrict__ A, const __bf16* __restrict__ W,
    const float* __restrict__ bias, float* __restrict__ Cf,
    __bf16* __restrict__ Cb, int M, int N, int K) {
  __shared__ alignas(16) __bf16 a_tile[2][128 * 32];
  __shared__ alignas(16) __bf16 b_tile[2][128 * 32];

  const int tid    = threadIdx.x;
  const int lane   = tid & 31;
  const int wave   = tid >> 5;
  const int tile_m = blockIdx.y * 128;
  const int tile_n = blockIdx.x * 128;
  const int m_off  = (wave & 3) * 32;
  const int n_off  = (wave >> 2) * 64;

  v8f acc[2][4];
#pragma unroll
  for (int a = 0; a < 2; ++a)
#pragma unroll
    for (int j = 0; j < 4; ++j) acc[a][j] = zero8();

  // staging map: 8-elem chunks; A and B each 512 chunks -> 2/thread each
  const int r0 = tid >> 2;              // 0..63
  const int r1 = r0 + 64;               // 64..127
  const int cc = (tid & 3) * 8;         // 0,8,16,24
  const __bf16* Ag = A + (size_t)tile_m * K;
  const __bf16* Bg = W + (size_t)tile_n * K;

  // prime buffer 0
  bf8v pa0 = *(const bf8v*)&Ag[(size_t)r0 * K + cc];
  bf8v pa1 = *(const bf8v*)&Ag[(size_t)r1 * K + cc];
  bf8v pb0 = *(const bf8v*)&Bg[(size_t)r0 * K + cc];
  bf8v pb1 = *(const bf8v*)&Bg[(size_t)r1 * K + cc];
  *(bf8v*)&a_tile[0][r0 * 32 + cc] = pa0;
  *(bf8v*)&a_tile[0][r1 * 32 + cc] = pa1;
  *(bf8v*)&b_tile[0][r0 * 32 + cc] = pb0;
  *(bf8v*)&b_tile[0][r1 * 32 + cc] = pb1;

  int cur = 0;
  for (int k0 = 0; k0 < K; k0 += 32) {
    __syncthreads();
    const bool more = (k0 + 32) < K;
    if (more) {  // prefetch next K slice into registers
      pa0 = *(const bf8v*)&Ag[(size_t)r0 * K + k0 + 32 + cc];
      pa1 = *(const bf8v*)&Ag[(size_t)r1 * K + k0 + 32 + cc];
      pb0 = *(const bf8v*)&Bg[(size_t)r0 * K + k0 + 32 + cc];
      pb1 = *(const bf8v*)&Bg[(size_t)r1 * K + k0 + 32 + cc];
    }
    if (k0 + 64 < K) {  // warm L2/WGP$ for the slice after that
      __builtin_prefetch(&Ag[(size_t)r0 * K + k0 + 64 + cc], 0, 3);
      __builtin_prefetch(&Bg[(size_t)r0 * K + k0 + 64 + cc], 0, 3);
    }
    v16bf af0 = load_frag(&a_tile[cur][m_off * 32], 32);
    v16bf af1 = load_frag(&a_tile[cur][(m_off + 16) * 32], 32);
    v16bf bf0 = load_frag(&b_tile[cur][n_off * 32], 32);
    v16bf bf1 = load_frag(&b_tile[cur][(n_off + 16) * 32], 32);
    v16bf bf2 = load_frag(&b_tile[cur][(n_off + 32) * 32], 32);
    v16bf bf3 = load_frag(&b_tile[cur][(n_off + 48) * 32], 32);
    acc[0][0] = wmma_bf16(af0, bf0, acc[0][0]);
    acc[1][0] = wmma_bf16(af1, bf0, acc[1][0]);
    acc[0][1] = wmma_bf16(af0, bf1, acc[0][1]);
    acc[1][1] = wmma_bf16(af1, bf1, acc[1][1]);
    acc[0][2] = wmma_bf16(af0, bf2, acc[0][2]);
    acc[1][2] = wmma_bf16(af1, bf2, acc[1][2]);
    acc[0][3] = wmma_bf16(af0, bf3, acc[0][3]);
    acc[1][3] = wmma_bf16(af1, bf3, acc[1][3]);
    if (more) {
      const int nxt = cur ^ 1;
      *(bf8v*)&a_tile[nxt][r0 * 32 + cc] = pa0;
      *(bf8v*)&a_tile[nxt][r1 * 32 + cc] = pa1;
      *(bf8v*)&b_tile[nxt][r0 * 32 + cc] = pb0;
      *(bf8v*)&b_tile[nxt][r1 * 32 + cc] = pb1;
    }
    cur ^= 1;
  }

  // C layout: VGPR r, lanes 0-15 -> M=r, lanes 16-31 -> M=r+8; N = lane&15.
  const int half = lane >> 4;
  const int col  = lane & 15;
#pragma unroll
  for (int a = 0; a < 2; ++a) {
#pragma unroll
    for (int j = 0; j < 4; ++j) {
#pragma unroll
      for (int r = 0; r < 8; ++r) {
        int gm  = tile_m + m_off + a * 16 + r + 8 * half;
        int gn  = tile_n + n_off + j * 16 + col;
        float v = acc[a][j][r] + bias[gn];
        if (EPI == 1) {
          v = 0.5f * v * (1.0f + erff(v * 0.70710678118654752f));  // exact GELU
          Cb[(size_t)gm * N + gn] = (__bf16)v;
        } else {
          Cf[(size_t)gm * N + gn] = v;
        }
      }
    }
  }
}

// ---------------------------------------------------------------------------
// Flash attention with additive param bias:
//   logits = (Q K^T + Pq Pk^T) / sqrt(hd),  softmax over t,  out = P V
// Q=K=V = src head slice; Pq=Pk = pe head slice.  One block per
// (b, h, 128-row q block); 8 wave32s, each wave owns 16 query rows.
// ---------------------------------------------------------------------------
__global__ __launch_bounds__(256) void attn_wmma(
    const __bf16* __restrict__ srcb,   // [B*S, 1024] bf16
    const __bf16* __restrict__ peb,    // [B*S, 1024] bf16
    float* __restrict__ outp) {        // [B*S, 1024] f32
  const int S = 1024, D = 1024, HD = 64;
  const int blk  = blockIdx.x;         // b*128 + h*8 + qb
  const int qb   = blk & 7;
  const int h    = (blk >> 3) & 15;
  const int b    = blk >> 7;
  const int q0   = qb * 128;
  const int tid  = threadIdx.x;
  const int lane = tid & 31;
  const int wave = tid >> 5;
  const int half = lane >> 4;
  const int col  = lane & 15;
  const size_t bS   = (size_t)b * S;
  const int    hcol = h * HD;
  const float  scale = 0.125f;         // 1/sqrt(64)

  __shared__ alignas(16) __bf16 kt[32 * 64];        // K tile  [t][hd]
  __shared__ alignas(16) __bf16 pkt[32 * 64];       // Pk tile [t][hd]
  __shared__ alignas(16) __bf16 vt[64 * 32];        // V^T tile [hd][t]
  __shared__ alignas(16) __bf16 pbuf[8][16 * 32];   // per-wave P tile [m][t]

  // Per-wave Q / Pq A-fragments (hd=64 -> two K=32 fragments each).
  const __bf16* qbase  = srcb + (bS + q0 + wave * 16) * D + hcol;
  const __bf16* pqbase = peb  + (bS + q0 + wave * 16) * D + hcol;
  v16bf qf0 = load_frag(qbase, D);
  v16bf qf1 = load_frag(qbase + 32, D);
  v16bf pf0 = load_frag(pqbase, D);
  v16bf pf1 = load_frag(pqbase + 32, D);

  v8f o0 = zero8(), o1 = zero8(), o2 = zero8(), o3 = zero8();
  float rmax[8], rsum[8];
#pragma unroll
  for (int r = 0; r < 8; ++r) { rmax[r] = -1e30f; rsum[r] = 0.f; }

  // staging map: 256 chunks of 8, one per thread
  const int sr = tid >> 3;            // 0..31 : t row
  const int sc = (tid & 7) * 8;       // 0..56 : hd col

  for (int t0 = 0; t0 < S; t0 += 32) {
    __syncthreads();
    {
      bf8v kv = *(const bf8v*)&srcb[(bS + t0 + sr) * D + hcol + sc];
      bf8v pk = *(const bf8v*)&peb[(bS + t0 + sr) * D + hcol + sc];
      *(bf8v*)&kt[sr * 64 + sc]  = kv;
      *(bf8v*)&pkt[sr * 64 + sc] = pk;
#pragma unroll
      for (int j = 0; j < 8; ++j) vt[(sc + j) * 32 + sr] = kv[j];  // V^T
    }
    __syncthreads();

    // logits tile (16 x 32): preload all B fragments, then the WMMA chain.
    v16bf bk0 = load_frag(&kt[0], 64);
    v16bf bk1 = load_frag(&kt[32], 64);
    v16bf bk2 = load_frag(&kt[16 * 64], 64);
    v16bf bk3 = load_frag(&kt[16 * 64 + 32], 64);
    v16bf bp0 = load_frag(&pkt[0], 64);
    v16bf bp1 = load_frag(&pkt[32], 64);
    v16bf bp2 = load_frag(&pkt[16 * 64], 64);
    v16bf bp3 = load_frag(&pkt[16 * 64 + 32], 64);
    v16bf vb0 = load_frag(&vt[0 * 16 * 32], 32);
    v16bf vb1 = load_frag(&vt[1 * 16 * 32], 32);
    v16bf vb2 = load_frag(&vt[2 * 16 * 32], 32);
    v16bf vb3 = load_frag(&vt[3 * 16 * 32], 32);

    v8f s0 = zero8(), s1 = zero8();
    s0 = wmma_bf16(qf0, bk0, s0);
    s1 = wmma_bf16(qf0, bk2, s1);
    s0 = wmma_bf16(qf1, bk1, s0);
    s1 = wmma_bf16(qf1, bk3, s1);
    s0 = wmma_bf16(pf0, bp0, s0);
    s1 = wmma_bf16(pf0, bp2, s1);
    s0 = wmma_bf16(pf1, bp1, s0);
    s1 = wmma_bf16(pf1, bp3, s1);
#pragma unroll
    for (int r = 0; r < 8; ++r) { s0[r] *= scale; s1[r] *= scale; }

    // --- online softmax (rows live in 16-lane halves; xor masks <16 keep
    //     the two halves independent) ---
    float tmax[8];
#pragma unroll
    for (int r = 0; r < 8; ++r) tmax[r] = fmaxf(s0[r], s1[r]);
#pragma unroll
    for (int m = 1; m < 16; m <<= 1)
#pragma unroll
      for (int r = 0; r < 8; ++r)
        tmax[r] = fmaxf(tmax[r], __shfl_xor(tmax[r], m, 32));

    float ps[8];
#pragma unroll
    for (int r = 0; r < 8; ++r) {
      float nm   = fmaxf(rmax[r], tmax[r]);
      float corr = __expf(rmax[r] - nm);
      rmax[r] = nm;
      s0[r] = __expf(s0[r] - nm);
      s1[r] = __expf(s1[r] - nm);
      rsum[r] *= corr;
      o0[r] *= corr; o1[r] *= corr; o2[r] *= corr; o3[r] *= corr;
      ps[r] = s0[r] + s1[r];
    }
#pragma unroll
    for (int m = 1; m < 16; m <<= 1)
#pragma unroll
      for (int r = 0; r < 8; ++r) ps[r] += __shfl_xor(ps[r], m, 32);
#pragma unroll
    for (int r = 0; r < 8; ++r) rsum[r] += ps[r];

    // C-layout -> A-layout for P via the wave-private LDS tile.  LDS ops from
    // one wave execute in order, so only a scheduling barrier is needed.
#pragma unroll
    for (int r = 0; r < 8; ++r) {
      pbuf[wave][(r + 8 * half) * 32 + col]      = (__bf16)s0[r];
      pbuf[wave][(r + 8 * half) * 32 + 16 + col] = (__bf16)s1[r];
    }
    __builtin_amdgcn_wave_barrier();

    v16bf af = load_frag(&pbuf[wave][0], 32);
    o0 = wmma_bf16(af, vb0, o0);
    o1 = wmma_bf16(af, vb1, o1);
    o2 = wmma_bf16(af, vb2, o2);
    o3 = wmma_bf16(af, vb3, o3);
  }

  // finalize: divide by row sums, scatter to [B*S, D] head slice
#pragma unroll
  for (int r = 0; r < 8; ++r) {
    float inv = 1.0f / rsum[r];
    int gm = q0 + wave * 16 + r + 8 * half;
    size_t base = (bS + gm) * (size_t)D + hcol;
    outp[base + col]      = o0[r] * inv;
    outp[base + 16 + col] = o1[r] * inv;
    outp[base + 32 + col] = o2[r] * inv;
    outp[base + 48 + col] = o3[r] * inv;
  }
}

// ---------------------------------------------------------------------------
// LayerNorm over D=1024 with optional residual add; optional f32/bf16 outputs.
// One block (256 threads) per token row; each thread owns 4 contiguous elems.
// Wave-level shfl reduction + a single cross-wave LDS combine.
// ---------------------------------------------------------------------------
__global__ __launch_bounds__(256) void ln_kernel(
    const float* __restrict__ a, const float* __restrict__ addv,
    const float* __restrict__ g, const float* __restrict__ be,
    float* __restrict__ outf, __bf16* __restrict__ outb) {
  const int D = 1024;
  const int row = blockIdx.x;
  const int i4  = threadIdx.x * 4;
  const int lane = threadIdx.x & 31;
  const int wave = threadIdx.x >> 5;
  const float* pa = a + (size_t)row * D;

  f4 v = *(const f4*)(pa + i4);
  if (addv) {
    f4 w = *(const f4*)(addv + (size_t)row * D + i4);
    v += w;
  }
  float s  = v[0] + v[1] + v[2] + v[3];
  float s2 = v[0] * v[0] + v[1] * v[1] + v[2] * v[2] + v[3] * v[3];

#pragma unroll
  for (int m = 1; m < 32; m <<= 1) {
    s  += __shfl_xor(s, m, 32);
    s2 += __shfl_xor(s2, m, 32);
  }
  __shared__ float2 red[8];
  if (lane == 0) red[wave] = make_float2(s, s2);
  __syncthreads();
  float ts = 0.f, ts2 = 0.f;
#pragma unroll
  for (int wv = 0; wv < 8; ++wv) { ts += red[wv].x; ts2 += red[wv].y; }

  const float mu   = ts * (1.0f / D);
  const float var  = ts2 * (1.0f / D) - mu * mu;
  const float rstd = rsqrtf(var + 1e-5f);

  f4 gg = *(const f4*)(g + i4);
  f4 bb = *(const f4*)(be + i4);
  f4 o;
#pragma unroll
  for (int j = 0; j < 4; ++j) o[j] = (v[j] - mu) * rstd * gg[j] + bb[j];
  if (outf) *(f4*)(outf + (size_t)row * D + i4) = o;
  if (outb) {
    bf4v ob;
#pragma unroll
    for (int j = 0; j < 4; ++j) ob[j] = (__bf16)o[j];
    *(bf4v*)(outb + (size_t)row * D + i4) = ob;
  }
}

// ---------------------------------------------------------------------------
// Host-side orchestration
// ---------------------------------------------------------------------------
extern "C" void kernel_launch(void* const* d_in, const int* in_sizes, int n_in,
                              void* d_out, int out_size, void* d_ws,
                              size_t ws_size, hipStream_t stream) {
  (void)in_sizes; (void)n_in; (void)out_size; (void)ws_size;
  const float* src  = (const float*)d_in[0];   // [4,1024,1024]
  const float* pv   = (const float*)d_in[1];   // [4,1024,4,768]
  const float* cW   = (const float*)d_in[2];   // [1024,768]
  const float* cb   = (const float*)d_in[3];   // [1024]
  const float* pe_g = (const float*)d_in[4];
  const float* pe_b = (const float*)d_in[5];
  const float* W1   = (const float*)d_in[6];   // [4096,1024]
  const float* b1   = (const float*)d_in[7];
  const float* W2   = (const float*)d_in[8];   // [1024,4096]
  const float* b2   = (const float*)d_in[9];
  const float* n1g  = (const float*)d_in[10];
  const float* n1b  = (const float*)d_in[11];
  const float* n2g  = (const float*)d_in[12];
  const float* n2b  = (const float*)d_in[13];
  float* out = (float*)d_out;

  // Workspace carve (256B aligned regions)
  char* w = (char*)d_ws;
  auto carve = [&](size_t bytes) -> char* {
    char* p = w;
    w += (bytes + 255) & ~(size_t)255;
    return p;
  };
  const size_t T = 4096;  // B*S tokens
  __bf16* src_bf = (__bf16*)carve(T * 1024 * 2);
  __bf16* pe_bf  = (__bf16*)carve(T * 1024 * 2);
  __bf16* x_bf   = (__bf16*)carve(T * 1024 * 2);
  __bf16* pvm_bf = (__bf16*)carve(T * 768 * 2);
  __bf16* cw_bf  = (__bf16*)carve((size_t)1024 * 768 * 2);
  __bf16* w1_bf  = (__bf16*)carve((size_t)4096 * 1024 * 2);
  __bf16* w2_bf  = (__bf16*)carve((size_t)1024 * 4096 * 2);
  __bf16* h_bf   = (__bf16*)carve(T * 4096 * 2);
  float*  pe_raw = (float*)carve(T * 1024 * 4);
  float*  attn_o = (float*)carve(T * 1024 * 4);
  float*  x_f    = (float*)carve(T * 1024 * 4);
  float*  ffn_f  = (float*)carve(T * 1024 * 4);

  // 1) precision converts / param mean (all 16B-vectorized)
  cvt_f32_bf16<<<512, 256, 0, stream>>>(src, src_bf, T * 1024 / 8);
  cvt_f32_bf16<<<256, 256, 0, stream>>>(cW, cw_bf, (size_t)1024 * 768 / 8);
  cvt_f32_bf16<<<512, 256, 0, stream>>>(W1, w1_bf, (size_t)4096 * 1024 / 8);
  cvt_f32_bf16<<<512, 256, 0, stream>>>(W2, w2_bf, (size_t)1024 * 4096 / 8);
  pv_mean_bf16<<<512, 256, 0, stream>>>(pv, pvm_bf);

  // 2) ParamEncoder GEMM: pe_raw = pvm @ char_W^T + char_b  (M=4096,N=1024,K=768)
  gemm_bf16_wmma<2><<<dim3(8, 32), 256, 0, stream>>>(
      pvm_bf, cw_bf, cb, pe_raw, nullptr, 4096, 1024, 768);

  // 3) LN(pe) -> bf16 head tensor for the bias GEMM
  ln_kernel<<<4096, 256, 0, stream>>>(pe_raw, nullptr, pe_g, pe_b, nullptr,
                                      pe_bf);

  // 4) fused flash attention + param bias  (B*H*(S/128) = 512 blocks)
  attn_wmma<<<512, 256, 0, stream>>>(src_bf, pe_bf, attn_o);

  // 5) x = LN(src + attn_out)
  ln_kernel<<<4096, 256, 0, stream>>>(attn_o, src, n1g, n1b, x_f, x_bf);

  // 6) h = gelu(x @ W1^T + b1)   (M=4096,N=4096,K=1024)
  gemm_bf16_wmma<1><<<dim3(32, 32), 256, 0, stream>>>(
      x_bf, w1_bf, b1, nullptr, h_bf, 4096, 4096, 1024);

  // 7) ffn = h @ W2^T + b2       (M=4096,N=1024,K=4096)
  gemm_bf16_wmma<2><<<dim3(8, 32), 256, 0, stream>>>(
      h_bf, w2_bf, b2, ffn_f, nullptr, 4096, 1024, 4096);

  // 8) out = LN(x + ffn)
  ln_kernel<<<4096, 256, 0, stream>>>(ffn_f, x_f, n2g, n2b, out, nullptr);
}
